// SparseSynthesisTransform_50852412785322
// MI455X (gfx1250) — matrix operand order repaired and checked by reading, not compiled, not run on hardware
//
#include <hip/hip_runtime.h>

// ---------------------------------------------------------------------------
// CDNA5 WMMA types (wave32, v_wmma_f32_16x16x32_f16)
// ---------------------------------------------------------------------------
typedef __attribute__((ext_vector_type(16))) _Float16 v16h;
typedef __attribute__((ext_vector_type(8)))  _Float16 v8h;
typedef __attribute__((ext_vector_type(8)))  float    v8f;
typedef __attribute__((ext_vector_type(4)))  unsigned vu4;

#define WAVES 2   // waves per block, one 16-voxel M tile each

union U8h { v8h h; vu4 u; };

__device__ __forceinline__ v8f wmma_f16(v16h a, v16h b, v8f c)
{
  return __builtin_amdgcn_wmma_f32_16x16x32_f16(false, a, false, b, (short)0, c,
                                                false, false);
}

template<int D> struct Log2    { static constexpr int v = 1 + Log2<D/2>::v; };
template<>      struct Log2<1> { static constexpr int v = 0; };

// ---------------------------------------------------------------------------
// Generic dense stride-1 SAME 3D conv, implicit GEMM on WMMA.
// Activations voxel-major f16: x (B, V, CIN). Weights (KT^3, CIN, COUT) f32.
// Optional f32 and/or f16 outputs (B, V, COUT).
// Per wave: M=16 voxels, N=16 couts, K over CIN in 32s, outer loop over taps
// with the tap's 16xCIN weight slice staged transposed (+pad) in LDS.
// Bounds handling: clamp address + 32-bit value selects; loads unpredicated.
// ---------------------------------------------------------------------------
template<int D, int KT, int CIN, int COUT, bool RELU, bool MASK, bool WF32, bool WF16>
__global__ void conv3_wmma(const _Float16* __restrict__ x, const float* __restrict__ W,
                           const float* __restrict__ bias, const float* __restrict__ mask,
                           float* __restrict__ out32, _Float16* __restrict__ out16)
{
  constexpr int V   = D * D * D;
  constexpr int LD  = Log2<D>::v;
  constexpr int KH  = KT / 2;
  constexpr int SWP = CIN + 8;                 // padded LDS row, halfs
  static_assert(CIN % 8 == 0, "CIN multiple of 8");
  __shared__ __align__(16) _Float16 sW[16 * SWP];

  const int lane = threadIdx.x & 31;
  const int mt   = blockIdx.x * WAVES + (threadIdx.x >> 5);
  const int nbase= blockIdx.y * 16;
  const int b    = blockIdx.z;
  const int n    = lane & 15;
  const int hi   = lane >> 4;
  const _Float16 hz = (_Float16)0.f;

  const int mvox = mt * 16 + (lane & 15);      // grid divides V exactly
  const int ox = mvox & (D - 1);
  const int oy = (mvox >> LD) & (D - 1);
  const int oz = mvox >> (2 * LD);

  v8f acc = {0.f, 0.f, 0.f, 0.f, 0.f, 0.f, 0.f, 0.f};

  for (int kd = 0; kd < KT; ++kd)
  for (int kh = 0; kh < KT; ++kh)
  for (int kw = 0; kw < KT; ++kw) {
    const int tap = (kd * KT + kh) * KT + kw;
    __syncthreads();
    for (int idx = threadIdx.x; idx < 16 * CIN; idx += blockDim.x) {
      const int nn = idx & 15, ci = idx >> 4;  // nn fastest -> coalesced W reads
      const float wv = (nbase + nn < COUT)
                         ? W[((size_t)tap * CIN + ci) * COUT + (nbase + nn)] : 0.f;
      sW[nn * SWP + ci] = (_Float16)wv;
    }
    __syncthreads();

    const int iz = oz + kd - KH, iy = oy + kh - KH, ix = ox + kw - KH;
    const bool valid = (unsigned)iz < (unsigned)D && (unsigned)iy < (unsigned)D
                    && (unsigned)ix < (unsigned)D;
    const int vsrc = valid ? ((iz * D + iy) * D + ix) : 0;   // clamp to voxel 0
    const v8h* __restrict__ xp = (const v8h*)(x + ((size_t)b * V + vsrc) * CIN);

    #pragma unroll
    for (int kc = 0; kc < CIN; kc += 32) {
      // --- A fragment: lane covers K groups {kb..kb+7, kb+16..kb+23}
      v16h a;
      #pragma unroll
      for (int g = 0; g < 2; ++g) {
        const int  kofs = kc + (hi ? 8 : 0) + g * 16;
        const bool vg   = valid && (kofs < CIN);
        U8h q; q.h = xp[(kofs < CIN) ? (kofs >> 3) : 0];
        #pragma unroll
        for (int t = 0; t < 4; ++t) q.u[t] = vg ? q.u[t] : 0u;  // 4x cndmask_b32
        #pragma unroll
        for (int j = 0; j < 8; ++j) a[g * 8 + j] = q.h[j];
      }
      // --- B fragment: lane holds column n; K from LDS as two 16B vectors
      const int  koff = kc + (hi ? 8 : 0);
      const bool bv0  = koff < CIN, bv1 = (koff + 16) < CIN;
      const v8h  b0   = *(const v8h*)(sW + n * SWP + (bv0 ? koff : 0));
      const v8h  b1   = *(const v8h*)(sW + n * SWP + (bv1 ? koff + 16 : 0));
      v16h bf;
      #pragma unroll
      for (int j = 0; j < 8; ++j) { bf[j] = bv0 ? b0[j] : hz; bf[8 + j] = bv1 ? b1[j] : hz; }
      acc = wmma_f16(a, bf, acc);
    }
  }

  if (nbase + n < COUT) {
    const float bv = bias[nbase + n];
    #pragma unroll
    for (int r = 0; r < 8; ++r) {
      const int ov = mt * 16 + r + (hi << 3);
      float val = acc[r] + bv;
      if (RELU) val = fmaxf(val, 0.f);
      if (MASK) val *= mask[(size_t)b * V + ov];
      const size_t oidx = ((size_t)b * V + ov) * COUT + (nbase + n);
      if constexpr (WF32) out32[oidx] = val;
      if constexpr (WF16) out16[oidx] = (_Float16)val;
    }
  }
}

// ---------------------------------------------------------------------------
// Generative transposed conv, kernel 5 stride 2, SAME (out = 2*in), gather
// form: out[o] += x[(o+2-k)/2] * W[k] for parity-matching in-range taps.
// Mask applied on store; writes f32 + f16 outputs.
// ---------------------------------------------------------------------------
template<int DIN, int CIN, int COUT>
__global__ void convt_wmma(const _Float16* __restrict__ x, const float* __restrict__ W,
                           const float* __restrict__ bias, const float* __restrict__ mask,
                           float* __restrict__ out32, _Float16* __restrict__ out16)
{
  constexpr int DO  = DIN * 2;
  constexpr int VO  = DO * DO * DO;
  constexpr int VI  = DIN * DIN * DIN;
  constexpr int LDO = Log2<DO>::v;
  constexpr int SWP = CIN + 8;
  __shared__ __align__(16) _Float16 sW[16 * SWP];

  const int lane = threadIdx.x & 31;
  const int mt   = blockIdx.x * WAVES + (threadIdx.x >> 5);
  const int nbase= blockIdx.y * 16;
  const int b    = blockIdx.z;
  const int n    = lane & 15;
  const int hi   = lane >> 4;
  const _Float16 hz = (_Float16)0.f;

  const int mvox = mt * 16 + (lane & 15);
  const int ox = mvox & (DO - 1);
  const int oy = (mvox >> LDO) & (DO - 1);
  const int oz = mvox >> (2 * LDO);

  v8f acc = {0.f, 0.f, 0.f, 0.f, 0.f, 0.f, 0.f, 0.f};

  for (int kd = 0; kd < 5; ++kd)
  for (int kh = 0; kh < 5; ++kh)
  for (int kw = 0; kw < 5; ++kw) {
    const int tap = (kd * 5 + kh) * 5 + kw;
    __syncthreads();
    for (int idx = threadIdx.x; idx < 16 * CIN; idx += blockDim.x) {
      const int nn = idx & 15, ci = idx >> 4;
      const float wv = (nbase + nn < COUT)
                         ? W[((size_t)tap * CIN + ci) * COUT + (nbase + nn)] : 0.f;
      sW[nn * SWP + ci] = (_Float16)wv;
    }
    __syncthreads();

    const int tz = oz + 2 - kd, ty = oy + 2 - kh, tx = ox + 2 - kw;
    const int iz = tz >> 1, iy = ty >> 1, ix = tx >> 1;
    const bool valid = !(tz & 1) && !(ty & 1) && !(tx & 1)
                     && (unsigned)iz < (unsigned)DIN
                     && (unsigned)iy < (unsigned)DIN
                     && (unsigned)ix < (unsigned)DIN;
    const int vsrc = valid ? ((iz * DIN + iy) * DIN + ix) : 0;
    const v8h* __restrict__ xp = (const v8h*)(x + ((size_t)b * VI + vsrc) * CIN);

    #pragma unroll
    for (int kc = 0; kc < CIN; kc += 32) {
      v16h a;
      #pragma unroll
      for (int g = 0; g < 2; ++g) {
        const int  kofs = kc + (hi ? 8 : 0) + g * 16;
        const bool vg   = valid && (kofs < CIN);
        U8h q; q.h = xp[(kofs < CIN) ? (kofs >> 3) : 0];
        #pragma unroll
        for (int t = 0; t < 4; ++t) q.u[t] = vg ? q.u[t] : 0u;
        #pragma unroll
        for (int j = 0; j < 8; ++j) a[g * 8 + j] = q.h[j];
      }
      const int  koff = kc + (hi ? 8 : 0);
      const bool bv0  = koff < CIN, bv1 = (koff + 16) < CIN;
      const v8h  b0   = *(const v8h*)(sW + n * SWP + (bv0 ? koff : 0));
      const v8h  b1   = *(const v8h*)(sW + n * SWP + (bv1 ? koff + 16 : 0));
      v16h bf;
      #pragma unroll
      for (int j = 0; j < 8; ++j) { bf[j] = bv0 ? b0[j] : hz; bf[8 + j] = bv1 ? b1[j] : hz; }
      acc = wmma_f16(a, bf, acc);
    }
  }

  if (nbase + n < COUT) {
    const float bv = bias[nbase + n];
    #pragma unroll
    for (int r = 0; r < 8; ++r) {
      const int ov = mt * 16 + r + (hi << 3);
      float val = (acc[r] + bv) * mask[(size_t)b * VO + ov];
      const size_t oidx = ((size_t)b * VO + ov) * COUT + (nbase + n);
      out32[oidx] = val;
      out16[oidx] = (_Float16)val;
    }
  }
}

// ---------------------------------------------------------------------------
// Inverse GDN as WMMA GEMM: norm = beta + (x*m)^2 . gamma^T  (M=vox,K=N=C),
// out = (x*m) * sqrt(norm). gamma^T staged once in LDS. f16 in -> f16 out.
// ---------------------------------------------------------------------------
template<int C, bool MASK>
__global__ void igdn_wmma(const _Float16* __restrict__ xin, _Float16* __restrict__ xout,
                          const float* __restrict__ beta, const float* __restrict__ gamma,
                          const float* __restrict__ mask, int V)
{
  static_assert(C % 32 == 0, "C multiple of 32");
  constexpr int SWP = C + 8;
  __shared__ __align__(16) _Float16 sG[16 * SWP];

  const int lane = threadIdx.x & 31;
  const int mt   = blockIdx.x * WAVES + (threadIdx.x >> 5);
  const int nbase= blockIdx.y * 16;
  const int b    = blockIdx.z;
  const int n    = lane & 15;
  const int hi   = lane >> 4;

  for (int idx = threadIdx.x; idx < 16 * C; idx += blockDim.x) {
    const int nn = idx & 15, kk = idx >> 4;
    sG[nn * SWP + kk] = (_Float16)gamma[(size_t)(nbase + nn) * C + kk];
  }
  __syncthreads();

  const int vrow = mt * 16 + (lane & 15);
  const _Float16 mh = MASK ? (_Float16)mask[(size_t)b * V + vrow] : (_Float16)1.f;
  const v8h* __restrict__ xp = (const v8h*)(xin + ((size_t)b * V + vrow) * C);

  v8f acc = {0.f, 0.f, 0.f, 0.f, 0.f, 0.f, 0.f, 0.f};
  #pragma unroll
  for (int kc = 0; kc < C; kc += 32) {
    v16h a;
    #pragma unroll
    for (int g = 0; g < 2; ++g) {
      const int kofs = kc + (hi ? 8 : 0) + g * 16;
      const v8h q = xp[kofs >> 3];
      #pragma unroll
      for (int j = 0; j < 8; ++j) {
        const _Float16 t = q[j] * mh;
        a[g * 8 + j] = t * t;
      }
    }
    const int koff = kc + (hi ? 8 : 0);
    const v8h  b0  = *(const v8h*)(sG + n * SWP + koff);
    const v8h  b1  = *(const v8h*)(sG + n * SWP + koff + 16);
    v16h bf;
    #pragma unroll
    for (int j = 0; j < 8; ++j) { bf[j] = b0[j]; bf[8 + j] = b1[j]; }
    acc = wmma_f16(a, bf, acc);
  }

  const float bt = beta[nbase + n];
  #pragma unroll
  for (int r = 0; r < 8; ++r) {
    const int ov = mt * 16 + r + (hi << 3);
    const float xv = (float)xin[((size_t)b * V + ov) * C + (nbase + n)];
    const float mo = MASK ? mask[(size_t)b * V + ov] : 1.f;
    xout[((size_t)b * V + ov) * C + (nbase + n)] = (_Float16)(xv * mo * sqrtf(bt + acc[r]));
  }
}

// ---------------------------------------------------------------------------
// (B,C,V) f32 -> (B,V,C) f16 transpose for the latent y.
// ---------------------------------------------------------------------------
__global__ void transpose_cv(const float* __restrict__ src, _Float16* __restrict__ dst,
                             int C, int V)
{
  const int idx = blockIdx.x * blockDim.x + threadIdx.x;
  const int b   = blockIdx.y;
  if (idx < C * V) {
    const int c = idx / V, v = idx % V;
    dst[((size_t)b * V + v) * C + c] = (_Float16)src[((size_t)b * C + c) * V + v];
  }
}

// ---------------------------------------------------------------------------
// Mask upsample: 5^3 stride-2 transpose support. min_ == nullptr -> all ones.
// ---------------------------------------------------------------------------
__global__ void upmask_k(const float* __restrict__ min_, float* __restrict__ mout, int Din)
{
  const int Do = Din * 2;
  const int Vo = Do * Do * Do;
  const int Vi = Din * Din * Din;
  const int b  = blockIdx.y;
  for (int o = blockIdx.x * blockDim.x + threadIdx.x; o < Vo;
       o += gridDim.x * blockDim.x) {
    const int ox = o % Do, oy = (o / Do) % Do, oz = o / (Do * Do);
    float r = 0.f;
    for (int kd = 0; kd < 5 && r == 0.f; ++kd) {
      const int tz = oz + 2 - kd; if (tz & 1) continue;
      const int iz = tz >> 1;     if (iz < 0 || iz >= Din) continue;
      for (int kh = 0; kh < 5 && r == 0.f; ++kh) {
        const int ty = oy + 2 - kh; if (ty & 1) continue;
        const int iy = ty >> 1;     if (iy < 0 || iy >= Din) continue;
        for (int kw = 0; kw < 5; ++kw) {
          const int tx = ox + 2 - kw; if (tx & 1) continue;
          const int ix = tx >> 1;     if (ix < 0 || ix >= Din) continue;
          if (!min_ || min_[(size_t)b * Vi + (iz * Din + iy) * Din + ix] > 0.f) { r = 1.f; break; }
        }
      }
    }
    mout[(size_t)b * Vo + o] = r;
  }
}

// ---------------------------------------------------------------------------
// Deterministic exact per-batch top-k: 8-pass byte-radix select over 64-bit
// key (monotone(f32) << 32) | ~index; mask := key >= kth-largest key.
// ---------------------------------------------------------------------------
struct TopkState { unsigned long long prefix; unsigned long long remaining; };

__device__ __forceinline__ unsigned long long tk_key(float f, int v)
{
  unsigned u = __float_as_uint(f);
  u = (u & 0x80000000u) ? ~u : (u | 0x80000000u);
  return ((unsigned long long)u << 32) | (unsigned)(0x7FFFFFFF - v);
}

__global__ void topk_init(TopkState* st, const int* kptr, unsigned* hist, int B_)
{
  for (int i = threadIdx.x; i < B_ * 256; i += blockDim.x) hist[i] = 0u;
  if ((int)threadIdx.x < B_) {
    st[threadIdx.x].prefix    = 0ull;
    st[threadIdx.x].remaining = (unsigned long long)(*kptr);
  }
}

__global__ void topk_hist(const float* __restrict__ p, const float* __restrict__ mask,
                          const TopkState* __restrict__ st, unsigned* __restrict__ hist,
                          int pass, int V)
{
  __shared__ unsigned lh[256];
  const int b = blockIdx.y;
  for (int i = threadIdx.x; i < 256; i += blockDim.x) lh[i] = 0u;
  __syncthreads();
  const int shift = 56 - 8 * pass;
  const unsigned long long pref = st[b].prefix;
  for (int v = blockIdx.x * blockDim.x + threadIdx.x; v < V;
       v += gridDim.x * blockDim.x) {
    if (mask[(size_t)b * V + v] > 0.f) {
      const unsigned long long k = tk_key(p[(size_t)b * V + v], v);
      const bool ok = (pass == 0) || ((k >> (shift + 8)) == (pref >> (shift + 8)));
      if (ok) atomicAdd(&lh[(unsigned)((k >> shift) & 0xFF)], 1u);
    }
  }
  __syncthreads();
  for (int i = threadIdx.x; i < 256; i += blockDim.x)
    if (lh[i]) atomicAdd(&hist[b * 256 + i], lh[i]);
}

__global__ void topk_scan(TopkState* st, unsigned* hist, int pass)
{
  const int b = blockIdx.x;
  if (threadIdx.x == 0) {
    const int shift = 56 - 8 * pass;
    const unsigned long long rem = st[b].remaining;
    unsigned long long cum = 0; int sel = 0;
    for (int bin = 255; bin >= 0; --bin) {
      const unsigned c = hist[b * 256 + bin];
      if (cum + c >= rem) { sel = bin; break; }
      cum += c;
    }
    st[b].prefix |= ((unsigned long long)sel) << shift;
    st[b].remaining = rem - cum;
  }
  __syncthreads();
  for (int i = threadIdx.x; i < 256; i += blockDim.x) hist[b * 256 + i] = 0u;
}

__global__ void topk_build(const float* __restrict__ p, float* __restrict__ mask,
                           const TopkState* __restrict__ st, int V)
{
  const int b = blockIdx.y;
  const unsigned long long T = st[b].prefix;
  for (int v = blockIdx.x * blockDim.x + threadIdx.x; v < V;
       v += gridDim.x * blockDim.x) {
    float nm = 0.f;
    if (mask[(size_t)b * V + v] > 0.f && tk_key(p[(size_t)b * V + v], v) >= T) nm = 1.f;
    mask[(size_t)b * V + v] = nm;
  }
}

// ---------------------------------------------------------------------------
// 1x1x1 color conv on f32 (B,V,C): out = (sum_ci (x*m)*W + b) * m.  NCDHW out.
// ---------------------------------------------------------------------------
__global__ void color_k(const float* __restrict__ x, const float* __restrict__ W,
                        const float* __restrict__ bias, const float* __restrict__ mask,
                        float* __restrict__ out, int B_, int Cin, int Cout, int V)
{
  const int idx = blockIdx.x * blockDim.x + threadIdx.x;
  const int tot = B_ * Cout * V;
  if (idx >= tot) return;
  const int v  = idx % V;
  const int co = (idx / V) % Cout;
  const int b  = idx / (V * Cout);
  const float m = mask[(size_t)b * V + v];
  float s = bias[co];
  for (int ci = 0; ci < Cin; ++ci)
    s += x[((size_t)b * V + v) * Cin + ci] * m * W[ci * Cout + co];
  out[((size_t)b * Cout + co) * V + v] = s * m;
}

// ---------------------------------------------------------------------------
// Host side
// ---------------------------------------------------------------------------
static inline dim3 conv_grid(int V, int Cout)
{
  return dim3((V / 16) / WAVES, (Cout + 15) / 16, 2);
}

static void launch_topk(const float* p, float* mask, const int* kptr, int V,
                        TopkState* st, unsigned* hist, int B, hipStream_t stream)
{
  topk_init<<<1, 256, 0, stream>>>(st, kptr, hist, B);
  int nb = (V + 255) / 256; if (nb > 256) nb = 256;
  for (int pass = 0; pass < 8; ++pass) {
    topk_hist<<<dim3(nb, B), 256, 0, stream>>>(p, mask, st, hist, pass, V);
    topk_scan<<<dim3(B), 256, 0, stream>>>(st, hist, pass);
  }
  topk_build<<<dim3(nb, B), 256, 0, stream>>>(p, mask, st, V);
}

extern "C" void kernel_launch(void* const* d_in, const int* in_sizes, int n_in,
                              void* d_out, int out_size, void* d_ws, size_t ws_size,
                              hipStream_t stream)
{
  (void)in_sizes; (void)n_in; (void)out_size; (void)ws_size;
  const float* y      = (const float*)d_in[0];
  const float* W_up1c = (const float*)d_in[1];  const float* b_up1c = (const float*)d_in[2];
  const float* g1_b   = (const float*)d_in[3];  const float* g1_g   = (const float*)d_in[4];
  const float* W_up1t = (const float*)d_in[5];  const float* b_up1t = (const float*)d_in[6];
  const float* g2_b   = (const float*)d_in[7];  const float* g2_g   = (const float*)d_in[8];
  const float* W_up2t = (const float*)d_in[9];  const float* b_up2t = (const float*)d_in[10];
  const float* g3_b   = (const float*)d_in[11]; const float* g3_g   = (const float*)d_in[12];
  const float* W_up3t = (const float*)d_in[13]; const float* b_up3t = (const float*)d_in[14];
  const float* W_col  = (const float*)d_in[15]; const float* b_col  = (const float*)d_in[16];
  const float* Wp1a = (const float*)d_in[17]; const float* bp1a = (const float*)d_in[18];
  const float* Wp1b = (const float*)d_in[19]; const float* bp1b = (const float*)d_in[20];
  const float* Wp2a = (const float*)d_in[21]; const float* bp2a = (const float*)d_in[22];
  const float* Wp2b = (const float*)d_in[23]; const float* bp2b = (const float*)d_in[24];
  const float* Wp3a = (const float*)d_in[25]; const float* bp3a = (const float*)d_in[26];
  const float* Wp3b = (const float*)d_in[27]; const float* bp3b = (const float*)d_in[28];
  const int* k1 = (const int*)d_in[29];
  const int* k2 = (const int*)d_in[30];
  const int* k3 = (const int*)d_in[31];

  const int B = 2;
  char* ws = (char*)d_ws;
  float*     bufA32 = (float*)(ws + 0);                  //  8 MB convT out f32 ping
  float*     bufB32 = (float*)(ws + (8ull  << 20));      //  8 MB convT out f32 pong
  _Float16*  bufA16 = (_Float16*)(ws + (16ull << 20));   //  4 MB convT out f16 ping
  _Float16*  bufB16 = (_Float16*)(ws + (20ull << 20));   //  4 MB convT out f16 pong
  _Float16*  bufG16 = (_Float16*)(ws + (24ull << 20));   //  4 MB igdn out f16
  _Float16*  bufH16 = (_Float16*)(ws + (28ull << 20));   // 16 MB predict hidden f16
  _Float16*  yT16   = (_Float16*)(ws + (44ull << 20));   // 256 KB latent (B,V,C) f16
  _Float16*  x8_16  = (_Float16*)(ws + (45ull << 20));   // 128 KB up1c out f16
  float*     bufP   = (float*)(ws + (46ull << 20));      //  2 MB predict logits f32
  float*     M1     = (float*)(ws + (48ull << 20));      //  2 MB mask ping
  float*     M2     = (float*)(ws + (50ull << 20));      //  2 MB mask pong
  unsigned*  hist   = (unsigned*)(ws + (52ull << 20));   //  2 KB
  TopkState* st     = (TopkState*)(ws + (52ull << 20) + 4096);
  float* out = (float*)d_out;

  // ---- latent transpose (B,128,512) f32 -> (B,512,128) f16
  transpose_cv<<<dim3(256, B), 256, 0, stream>>>(y, yT16, 128, 512);

  // ---- up_1: conv5 128->64 @8^3, IGDN, upmask, convT 64->32 -> 16^3
  conv3_wmma<8, 5, 128, 64, false, false, false, true>
      <<<conv_grid(512, 64), 32 * WAVES, 0, stream>>>(yT16, W_up1c, b_up1c, nullptr, nullptr, x8_16);
  igdn_wmma<64, false>
      <<<dim3((512 / 16) / WAVES, 4, B), 32 * WAVES, 0, stream>>>(x8_16, bufG16, g1_b, g1_g, nullptr, 512);
  upmask_k<<<dim3(16, B), 256, 0, stream>>>(nullptr, M1, 8);
  convt_wmma<8, 64, 32>
      <<<conv_grid(4096, 32), 32 * WAVES, 0, stream>>>(bufG16, W_up1t, b_up1t, M1, bufB32, bufB16);
  // predict + prune @16^3
  conv3_wmma<16, 3, 32, 16, true, true, false, true>
      <<<conv_grid(4096, 16), 32 * WAVES, 0, stream>>>(bufB16, Wp1a, bp1a, M1, nullptr, bufH16);
  conv3_wmma<16, 3, 16, 1, false, true, true, false>
      <<<conv_grid(4096, 1), 32 * WAVES, 0, stream>>>(bufH16, Wp1b, bp1b, M1, bufP, nullptr);
  launch_topk(bufP, M1, k1, 4096, st, hist, B, stream);

  // ---- up_2: IGDN (masked), upmask, convT 32->32 -> 32^3
  igdn_wmma<32, true>
      <<<dim3((4096 / 16) / WAVES, 2, B), 32 * WAVES, 0, stream>>>(bufB16, bufG16, g2_b, g2_g, M1, 4096);
  upmask_k<<<dim3(128, B), 256, 0, stream>>>(M1, M2, 16);
  convt_wmma<16, 32, 32>
      <<<conv_grid(32768, 32), 32 * WAVES, 0, stream>>>(bufG16, W_up2t, b_up2t, M2, bufA32, bufA16);
  // predict + prune @32^3
  conv3_wmma<32, 3, 32, 16, true, true, false, true>
      <<<conv_grid(32768, 16), 32 * WAVES, 0, stream>>>(bufA16, Wp2a, bp2a, M2, nullptr, bufH16);
  conv3_wmma<32, 3, 16, 1, false, true, true, false>
      <<<conv_grid(32768, 1), 32 * WAVES, 0, stream>>>(bufH16, Wp2b, bp2b, M2, bufP, nullptr);
  launch_topk(bufP, M2, k2, 32768, st, hist, B, stream);

  // ---- up_3: IGDN (masked), upmask, convT 32->8 -> 64^3
  igdn_wmma<32, true>
      <<<dim3((32768 / 16) / WAVES, 2, B), 32 * WAVES, 0, stream>>>(bufA16, bufG16, g3_b, g3_g, M2, 32768);
  upmask_k<<<dim3(1024, B), 256, 0, stream>>>(M2, M1, 32);
  convt_wmma<32, 32, 8>
      <<<conv_grid(262144, 8), 32 * WAVES, 0, stream>>>(bufG16, W_up3t, b_up3t, M1, bufB32, bufB16);
  // predict + prune @64^3
  conv3_wmma<64, 3, 8, 16, true, true, false, true>
      <<<conv_grid(262144, 16), 32 * WAVES, 0, stream>>>(bufB16, Wp3a, bp3a, M1, nullptr, bufH16);
  conv3_wmma<64, 3, 16, 1, false, true, true, false>
      <<<conv_grid(262144, 1), 32 * WAVES, 0, stream>>>(bufH16, Wp3b, bp3b, M1, bufP, nullptr);
  launch_topk(bufP, M1, k3, 262144, st, hist, B, stream);

  // ---- color 1x1x1 conv 8->3 -> d_out (NCDHW)
  const int tot = B * 3 * 262144;
  color_k<<<(tot + 255) / 256, 256, 0, stream>>>(
      bufB32, W_col, b_col, M1, out, B, 8, 3, 262144);
}
